// DynamicRNN_28552942584268
// MI455X (gfx1250) — matrix operand order repaired and checked by reading, not compiled, not run on hardware
//
#include <hip/hip_runtime.h>

typedef __attribute__((ext_vector_type(16))) __bf16 v16bf;
typedef __attribute__((ext_vector_type(8)))  __bf16 v8bf;
typedef __attribute__((ext_vector_type(8)))  float  v8f;
typedef __attribute__((ext_vector_type(4)))  float  v4f;

#define TT  512
#define BB  2048
#define HH  64
#define LDW 192   // leading dim of Wx/Wh (3*H)

static __device__ __forceinline__ v8f vbcast8(float v){
  v8f r;
#pragma unroll
  for (int j = 0; j < 8; ++j) r[j] = v;
  return r;
}

// Build one 32x16 bf16 B-fragment from a row-major (64 x 192) f32 weight matrix.
// Tile columns [nc, nc+16), K-chunk [kc, kc+32).
// Layout (this code's contiguous k order, not the ISA table): lanes 0-15 hold K=kc..kc+15, lanes 16-31 hold K=kc+16..kc+31;
// VGPR j packs rows K=base+2j (lo16) and K=base+2j+1 (hi16), N = lane%16.
static __device__ __forceinline__ v16bf load_wfrag(const float* __restrict__ W,
                                                   int nc, int kc, int lane){
  const int n  = nc + (lane & 15);
  const int kb = kc + ((lane >> 4) << 4);
  v16bf f;
#pragma unroll
  for (int j = 0; j < 8; ++j){
    f[2*j]   = (__bf16)W[(size_t)(kb + 2*j    ) * LDW + n];
    f[2*j+1] = (__bf16)W[(size_t)(kb + 2*j + 1) * LDW + n];
  }
  return f;
}

// Build one 16x32 bf16 A-fragment from a row-major 16x64 bf16 LDS buffer.
// Layout (matches B fragment): lanes 0-15 (M=lane) hold K=kc+0..15;
// lanes 16-31 (M=lane-16) hold K=kc+16..31.
static __device__ __forceinline__ v16bf lds_afrag(const __bf16* buf, int m,
                                                  int kc, int lhalf){
  const int koff = kc + (lhalf << 4);
  v8bf lo = *(const v8bf*)(buf + m * 64 + koff);        // ds_load_b128
  v8bf hi = *(const v8bf*)(buf + m * 64 + koff + 8);    // ds_load_b128
  v16bf f;
#pragma unroll
  for (int j = 0; j < 8; ++j){ f[j] = lo[j]; f[8 + j] = hi[j]; }
  return f;
}

static __device__ __forceinline__ v8f wmma_bf16(v16bf a, v16bf b, v8f c){
  // (neg_a, A, neg_b, B, c_mod, C, reuse_a, reuse_b)
  return __builtin_amdgcn_wmma_f32_16x16x32_bf16(false, a, false, b,
                                                 (short)0, c, false, false);
}

// Fast activations: single v_exp_f32 + v_rcp_f32, no IEEE divide expansion.
static __device__ __forceinline__ float fast_sigmoid(float v){
  return __builtin_amdgcn_rcpf(1.0f + __expf(-v));
}
static __device__ __forceinline__ float fast_tanh(float v){
  // stable at both tails: 1 - 2/(e^{2v}+1); rcp(inf)=0 -> +1, rcp(1)=1 -> -1
  float e = __expf(2.0f * v);
  return 1.0f - 2.0f * __builtin_amdgcn_rcpf(e + 1.0f);
}

// Cooperative load of the 16x64 f32 x-tile for timestep t -> bf16 in LDS.
// 128 threads x 8 floats = 1024 elements. Non-temporal: x is streamed once.
static __device__ __forceinline__ void prefetch_x(const float* __restrict__ x,
                                                  int t, int rowbase, int tid,
                                                  __bf16* dst){
  const int row = tid >> 3;          // 0..15
  const int c8  = (tid & 7) << 3;    // 0,8,...,56
  const float* src = x + (((size_t)(rowbase + row)) * TT + t) * 64 + c8;
  v4f a = __builtin_nontemporal_load((const v4f*)src);
  v4f c = __builtin_nontemporal_load((const v4f*)(src + 4));
  v8bf o;
#pragma unroll
  for (int j = 0; j < 4; ++j){ o[j] = (__bf16)a[j]; o[4 + j] = (__bf16)c[j]; }
  *(v8bf*)(dst + row * 64 + c8) = o;  // ds_store_b128
}

// One workgroup = 16 batch rows, 4 waves; wave w owns columns [16w,16w+16) of
// each gate. Fused xg = x@Wx + b; serial scan over T with LDS-resident h.
__global__ __launch_bounds__(128) void DynamicRNN_gru_scan(
    const float* __restrict__ x,   // (B, T, 64)
    const float* __restrict__ Wx,  // (64, 192)
    const float* __restrict__ Wh,  // (64, 192)
    const float* __restrict__ bias,// (192,)
    float* __restrict__ out,       // (B, T, 64)
    float* __restrict__ hfin)      // (B, 64)
{
  __shared__ __bf16 hbuf [16 * 64];
  __shared__ __bf16 rhbuf[16 * 64];
  __shared__ __bf16 xbuf [2][16 * 64];

  const int tid     = threadIdx.x;
  const int wv      = tid >> 5;
  const int ln      = tid & 31;
  const int lhalf   = ln >> 4;   // which 16-lane half
  const int lmod    = ln & 15;   // N (column within tile) for B/C/D
  const int rowbase = blockIdx.x * 16;

  const int zc = 16 * wv, rc = 64 + 16 * wv, hc = 128 + 16 * wv;

  // Constant B-fragments, held in VGPRs for the whole scan (12 x 8 VGPRs).
  v16bf WXZ0 = load_wfrag(Wx, zc, 0, ln), WXZ1 = load_wfrag(Wx, zc, 32, ln);
  v16bf WXR0 = load_wfrag(Wx, rc, 0, ln), WXR1 = load_wfrag(Wx, rc, 32, ln);
  v16bf WXH0 = load_wfrag(Wx, hc, 0, ln), WXH1 = load_wfrag(Wx, hc, 32, ln);
  v16bf UZ0  = load_wfrag(Wh, zc, 0, ln), UZ1  = load_wfrag(Wh, zc, 32, ln);
  v16bf UR0  = load_wfrag(Wh, rc, 0, ln), UR1  = load_wfrag(Wh, rc, 32, ln);
  v16bf UH0  = load_wfrag(Wh, hc, 0, ln), UH1  = load_wfrag(Wh, hc, 32, ln);
  const float bz = bias[zc + lmod];
  const float br = bias[rc + lmod];
  const float bh = bias[hc + lmod];

  // h = 0 in LDS (each of 128 threads zeroes 8 bf16)
  {
    v8bf z8;
#pragma unroll
    for (int j = 0; j < 8; ++j) z8[j] = (__bf16)0.0f;
    *(v8bf*)(&hbuf[tid * 8]) = z8;
  }
  float hown[8];
#pragma unroll
  for (int j = 0; j < 8; ++j) hown[j] = 0.0f;

  prefetch_x(x, 0, rowbase, tid, xbuf[0]);
  __syncthreads();

  for (int t = 0; t < TT; ++t){
    const __bf16* xb = xbuf[t & 1];

    // A-fragments: x_t and h (full K=64, two 32-chunks each)
    v16bf xa0 = lds_afrag(xb,   lmod, 0,  lhalf);
    v16bf xa1 = lds_afrag(xb,   lmod, 32, lhalf);
    v16bf ha0 = lds_afrag(hbuf, lmod, 0,  lhalf);
    v16bf ha1 = lds_afrag(hbuf, lmod, 32, lhalf);

    // z_pre = b_z + x@Wxz + h@Uz   (f32 accumulate)
    v8f zacc = vbcast8(bz);
    zacc = wmma_bf16(xa0, WXZ0, zacc);
    zacc = wmma_bf16(xa1, WXZ1, zacc);
    zacc = wmma_bf16(ha0, UZ0,  zacc);
    zacc = wmma_bf16(ha1, UZ1,  zacc);

    // r_pre = b_r + x@Wxr + h@Ur
    v8f racc = vbcast8(br);
    racc = wmma_bf16(xa0, WXR0, racc);
    racc = wmma_bf16(xa1, WXR1, racc);
    racc = wmma_bf16(ha0, UR0,  racc);
    racc = wmma_bf16(ha1, UR1,  racc);

    // Prefetch next timestep's x tile into the other LDS buffer.
    if (t + 1 < TT) prefetch_x(x, t + 1, rowbase, tid, xbuf[(t + 1) & 1]);

    float zg[8], rg[8];
#pragma unroll
    for (int j = 0; j < 8; ++j){
      zg[j] = fast_sigmoid(zacc[j]);
      rg[j] = fast_sigmoid(racc[j]);
    }
    // rh = r * h  (C-layout: element j -> row m=j+8*lhalf, col 16w+lmod)
#pragma unroll
    for (int j = 0; j < 8; ++j){
      rhbuf[(j + 8 * lhalf) * 64 + 16 * wv + lmod] = (__bf16)(rg[j] * hown[j]);
    }
    __syncthreads();

    // hh_pre = b_h + x@Wxh + (r*h)@Uh
    v16bf ra0 = lds_afrag(rhbuf, lmod, 0,  lhalf);
    v16bf ra1 = lds_afrag(rhbuf, lmod, 32, lhalf);
    v8f hacc = vbcast8(bh);
    hacc = wmma_bf16(xa0, WXH0, hacc);
    hacc = wmma_bf16(xa1, WXH1, hacc);
    hacc = wmma_bf16(ra0, UH0,  hacc);
    hacc = wmma_bf16(ra1, UH1,  hacc);

#pragma unroll
    for (int j = 0; j < 8; ++j){
      const float hh = fast_tanh(hacc[j]);
      const float o  = zg[j] * hown[j] + (1.0f - zg[j]) * hh;
      const int m = j + 8 * lhalf;
      // out is write-once streaming traffic: keep it non-temporal.
      __builtin_nontemporal_store(
          o, out + (((size_t)(rowbase + m)) * TT + t) * 64 + 16 * wv + lmod);
      hown[j] += o;                                   // h_sum carry
      hbuf[m * 64 + 16 * wv + lmod] = (__bf16)hown[j];
    }
    __syncthreads();
  }

  // final_state = h_sum after last step
#pragma unroll
  for (int j = 0; j < 8; ++j){
    const int m = j + 8 * lhalf;
    hfin[(size_t)(rowbase + m) * 64 + 16 * wv + lmod] = hown[j];
  }
}

extern "C" void kernel_launch(void* const* d_in, const int* in_sizes, int n_in,
                              void* d_out, int out_size, void* d_ws, size_t ws_size,
                              hipStream_t stream) {
  (void)in_sizes; (void)n_in; (void)d_ws; (void)ws_size; (void)out_size;
  const float* x    = (const float*)d_in[0];  // (2048, 512, 64)
  const float* Wx   = (const float*)d_in[1];  // (64, 192)
  const float* Wh   = (const float*)d_in[2];  // (64, 192)
  const float* bias = (const float*)d_in[3];  // (192,)
  float* out  = (float*)d_out;                              // (B,T,H)
  float* hfin = out + (size_t)BB * TT * HH;                 // (B,H) appended
  DynamicRNN_gru_scan<<<BB / 16, 128, 0, stream>>>(x, Wx, Wh, bias, out, hfin);
}
